// WaveNet_22608707846873
// MI455X (gfx1250) — matrix operand order repaired and checked by reading, not compiled, not run on hardware
//
#include <hip/hip_runtime.h>
#include <math.h>

typedef __attribute__((ext_vector_type(16))) _Float16 v16h;
typedef __attribute__((ext_vector_type(8)))  _Float16 v8h;
typedef __attribute__((ext_vector_type(4)))  _Float16 v4h;
typedef __attribute__((ext_vector_type(2)))  _Float16 v2h;
typedef __attribute__((ext_vector_type(8)))  float    v8f;
typedef __attribute__((ext_vector_type(4)))  float    v4f;

#define T_LEN 32000
#define TF    200
#define NF    26
#define HID4  2048
#define HIDN  512
#define RC    64
#define CH    128   // 2*RC
#define CCH   128   // CC
#define SCH   256   // SC
#define OCH   256   // OC
#define KTOT  256   // RC + RC + CC
#define NLAY  16
#define NT    32    // time columns per layer block

__device__ inline _Float16 f2h(float f) { return (_Float16)f; }
__device__ inline float sigmoidf_(float x) { return 1.0f / (1.0f + __expf(-x)); }

// ============ fragment layouts (ISA 7.12.2, wave32) =========================
// A fragment (16x32 f16): lane holds row (lane&15); k = (lane>>4)*16 + e.
// B fragment (32x16 f16): lane holds col (lane&15); k = (lane>>4)*16 + e.
// Both stored pre-swizzled: tile -> lane -> 16 contiguous halfs (32B per lane).

__device__ inline int a_swz_idx(int nkt, int row, int k) {
  int tile = (row >> 4) * nkt + (k >> 5);
  int kl   = k & 31;
  int b1   = (kl >> 4) & 1;
  int rem  = kl & 15;
  int lane = (row & 15) | (b1 << 4);
  int e    = rem;
  return (tile * 32 + lane) * 16 + e;
}

__device__ inline v16h load_frag_a(const _Float16* __restrict__ Wg, int tile) {
  const _Float16* p = Wg + (tile * 32 + (threadIdx.x & 31)) * 16;
  v8h lo = *(const v8h*)(p);
  v8h hi = *(const v8h*)(p + 8);
  return __builtin_shufflevector(lo, hi, 0, 1, 2, 3, 4, 5, 6, 7,
                                 8, 9, 10, 11, 12, 13, 14, 15);
}

__device__ inline void store_b_swz(_Float16* X, int ntn, int k, int n, _Float16 v) {
  int tile = (k >> 5) * ntn + (n >> 4);
  int lane = (n & 15) | (((k >> 4) & 1) << 4);
  X[(tile * 32 + lane) * 16 + (k & 15)] = v;
}

__device__ inline v16h load_frag_b(const _Float16* X, int tile) {
  const _Float16* p = X + (tile * 32 + (threadIdx.x & 31)) * 16;
  v8h lo = *(const v8h*)(p);
  v8h hi = *(const v8h*)(p + 8);
  return __builtin_shufflevector(lo, hi, 0, 1, 2, 3, 4, 5, 6, 7,
                                 8, 9, 10, 11, 12, 13, 14, 15);
}

__device__ inline v8f wmma_f16(v16h a, v16h b, v8f c) {
  return __builtin_amdgcn_wmma_f32_16x16x32_f16(false, a, false, b, (short)0, c, false, false);
}

// ============ weight pre-pack (once per call; L2-resident thereafter) =======
__global__ __launch_bounds__(256)
void pack_weights_kernel(const float* __restrict__ dil_w, const float* __restrict__ cond_w,
                         const float* __restrict__ skip_w, const float* __restrict__ res_w,
                         _Float16* __restrict__ W1p, _Float16* __restrict__ SWp,
                         _Float16* __restrict__ RWp)
{
  const int layer = blockIdx.y;
  const float* dw = dil_w  + (size_t)layer * CH * RC * 2;
  const float* cw = cond_w + (size_t)layer * CH * CCH;
  const float* sw = skip_w + (size_t)layer * SCH * RC;
  const float* rw = res_w  + (size_t)layer * RC * RC;
  _Float16* w1 = W1p + (size_t)layer * CH * KTOT;
  _Float16* sp = SWp + (size_t)layer * SCH * RC;
  _Float16* rp = RWp + (size_t)layer * RC * RC;

  int idx = blockIdx.x * 256 + threadIdx.x;
  const int N1 = CH * KTOT, N2 = SCH * RC, N3 = RC * RC;
  if (idx < N1) {
    int o = idx / KTOT, k = idx % KTOT;
    float w;
    if (k < RC)          w = dw[(o * RC + k) * 2];
    else if (k < 2 * RC) w = dw[(o * RC + (k - RC)) * 2 + 1];
    else                 w = cw[o * CCH + (k - 2 * RC)];
    w1[a_swz_idx(8, o, k)] = f2h(w);
  } else if (idx < N1 + N2) {
    int j = idx - N1, o = j / RC, k = j % RC;
    sp[a_swz_idx(2, o, k)] = f2h(sw[j]);
  } else if (idx < N1 + N2 + N3) {
    int j = idx - N1 - N2, o = j / RC, k = j % RC;
    rp[a_swz_idx(2, o, k)] = f2h(rw[j]);
  }
}

__global__ __launch_bounds__(256)
void pack_head_kernel(const float* __restrict__ out1_w, const float* __restrict__ out2_w,
                      _Float16* __restrict__ O1p, _Float16* __restrict__ O2p)
{
  int idx = blockIdx.x * 256 + threadIdx.x;
  if (idx >= 2 * SCH * SCH) return;
  const float* src = (idx < SCH * SCH) ? out1_w : out2_w;
  _Float16* dst    = (idx < SCH * SCH) ? O1p : O2p;
  int j = idx & (SCH * SCH - 1);
  int o = j / SCH, k = j % SCH;
  dst[a_swz_idx(8, o, k)] = f2h(src[j]);
}

// ---------------- Bi-LSTM (sequential, latency-bound; LDS-resident state) ----
__global__ __launch_bounds__(1024)
void lstm_kernel(const float* __restrict__ mfcc,
                 const float* __restrict__ wih_f, const float* __restrict__ whh_f, const float* __restrict__ b_f,
                 const float* __restrict__ wih_r, const float* __restrict__ whh_r, const float* __restrict__ b_r,
                 float* __restrict__ hf, float* __restrict__ hr)
{
  __shared__ float h[HIDN], cst[HIDN], gates[HID4], zin[NF];
  const int tid = threadIdx.x;
  const int batch = blockIdx.x >> 1;
  const int dir   = blockIdx.x & 1;
  const float* wih = dir ? wih_r : wih_f;
  const float* whh = dir ? whh_r : whh_f;
  const float* bb  = dir ? b_r   : b_f;
  float* hout = dir ? hr : hf;

  for (int i = tid; i < HIDN; i += 1024) { h[i] = 0.0f; cst[i] = 0.0f; }
  __syncthreads();

  for (int s = 0; s < TF; ++s) {
    int tt = dir ? (TF - 1 - s) : s;
    if (tid < NF) zin[tid] = mfcc[(batch * TF + tt) * NF + tid];
    __syncthreads();
    for (int g = tid; g < HID4; g += 1024) {
      float acc = bb[g];
      const float* wi = wih + g * NF;
#pragma unroll
      for (int f = 0; f < NF; ++f) acc += zin[f] * wi[f];
      const float* wh = whh + g * HIDN;
      for (int j = 0; j < HIDN; ++j) acc += h[j] * wh[j];
      gates[g] = acc;
    }
    __syncthreads();
    if (tid < HIDN) {
      float ig = sigmoidf_(gates[tid]);
      float fg = sigmoidf_(gates[HIDN + tid]);
      float gg = tanhf(gates[2 * HIDN + tid]);
      float og = sigmoidf_(gates[3 * HIDN + tid]);
      float c2 = fg * cst[tid] + ig * gg;
      cst[tid] = c2;
      float hh = og * tanhf(c2);
      h[tid] = hh;
      hout[(batch * TF + tt) * HIDN + tid] = hh;
    }
    __syncthreads();
  }
}

// ---------------- projection ------------------------------------------------
__global__ __launch_bounds__(128)
void proj_kernel(const float* __restrict__ hf, const float* __restrict__ hr,
                 const float* __restrict__ proj_w, const float* __restrict__ proj_b,
                 float* __restrict__ c)
{
  __shared__ float hc[2 * HIDN];
  const int tid = threadIdx.x;
  const int b = blockIdx.x / TF;
  const int t = blockIdx.x % TF;
  for (int i = tid; i < HIDN; i += 128) {
    hc[i]        = hf[(b * TF + t) * HIDN + i];
    hc[HIDN + i] = hr[(b * TF + t) * HIDN + i];
  }
  __syncthreads();
  float acc = proj_b[tid];
  const float* w = proj_w + tid * (2 * HIDN);
  for (int j = 0; j < 2 * HIDN; ++j) acc += hc[j] * w[j];
  c[(b * CCH + tid) * TF + t] = acc;
}

// ---------------- upsample path ---------------------------------------------
__global__ void upw_transpose_kernel(const float* __restrict__ up_w, float* __restrict__ up_wT)
{
  int idx = blockIdx.x * blockDim.x + threadIdx.x;
  int total = CCH * CCH * 320;
  if (idx >= total) return;
  int k = idx % 320;
  int o = (idx / 320) % CCH;
  int i = idx / (320 * CCH);
  up_wT[(k * CCH + o) * CCH + i] = up_w[idx];
}

__global__ __launch_bounds__(128)
void upsample_kernel(const float* __restrict__ c, const float* __restrict__ up_wT,
                     const float* __restrict__ up_b, float* __restrict__ cond)
{
  __shared__ float c0[CCH], c1[CCH];
  const int tid = threadIdx.x;
  const int b = blockIdx.x / 160;
  const int r = blockIdx.x % 160;
  const int k0 = (r <= 79) ? (79 - r) : (239 - r);
  const int tau0_off = (r + k0 - 239) / 160;  // exactly -1 or 0
  const float* w0 = up_wT + (size_t)(319 - k0) * CCH * CCH + tid * CCH;
  const float* w1 = up_wT + (size_t)(319 - (k0 + 160)) * CCH * CCH + tid * CCH;
  const float bo = up_b[tid];
  for (int s = 0; s < TF; ++s) {
    int t = r + 160 * s;
    int tau0 = s + tau0_off;
    int tau1 = tau0 + 1;
    c0[tid] = (tau0 >= 0 && tau0 < TF) ? c[(b * CCH + tid) * TF + tau0] : 0.0f;
    c1[tid] = (tau1 >= 0 && tau1 < TF) ? c[(b * CCH + tid) * TF + tau1] : 0.0f;
    __syncthreads();
    float acc = bo;
    for (int i = 0; i < CCH; ++i) acc += c0[i] * w0[i] + c1[i] * w1[i];
    cond[(size_t)(b * CCH + tid) * T_LEN + t] = acc;
    __syncthreads();
  }
}

// ---------------- x init ----------------------------------------------------
__global__ void xinit_kernel(const float* __restrict__ audio, const float* __restrict__ in_w,
                             const float* __restrict__ in_b, float* __restrict__ x)
{
  int idx = blockIdx.x * blockDim.x + threadIdx.x;
  int total = 2 * RC * T_LEN;
  if (idx >= total) return;
  int t = idx % T_LEN;
  int rch = (idx / T_LEN) % RC;
  int b = idx / (RC * T_LEN);
  x[idx] = in_w[rch] * audio[b * T_LEN + t] + in_b[rch];
}

// ---------------- WaveNet layer: barrier-free WMMA k-loops ------------------
__global__ __launch_bounds__(256)
void layer_kernel(const float* __restrict__ x_in, float* __restrict__ x_out,
                  const float* __restrict__ cond,
                  const _Float16* __restrict__ W1p, const float* __restrict__ dil_b,
                  const float* __restrict__ cond_b,
                  const _Float16* __restrict__ SWp, const float* __restrict__ skip_b,
                  const _Float16* __restrict__ RWp, const float* __restrict__ res_b,
                  float* __restrict__ skip_sum, int dil, int first)
{
  __shared__ __align__(16) _Float16 Xin[KTOT * NT];  // 16KB, B-swizzled (ntn=2)
  __shared__ float    H[CH * NT];                    // 16KB pre-activation
  __shared__ __align__(16) _Float16 G[RC * NT];      // 4KB gated, B-swizzled (ntn=2)
  __shared__ float hb[CH], sb[SCH], rb[RC];

  const int tid = threadIdx.x;
  const int nTB = T_LEN / NT;
  const int bt = blockIdx.x / nTB;
  const int t0 = (blockIdx.x % nTB) * NT;
  const float* xb = x_in + (size_t)bt * RC * T_LEN;
  const float* cb = cond + (size_t)bt * CCH * T_LEN;

  for (int i = tid; i < CH;  i += 256) hb[i] = dil_b[i] + cond_b[i];
  for (int i = tid; i < SCH; i += 256) sb[i] = skip_b[i];
  for (int i = tid; i < RC;  i += 256) rb[i] = res_b[i];

  // shifted x(t-d): scalar (left boundary + d-unaligned addresses)
  for (int id = tid; id < RC * NT; id += 256) {
    int k = id >> 5, n = id & 31;
    int ts = t0 + n - dil;
    store_b_swz(Xin, 2, k, n, f2h(ts >= 0 ? xb[(size_t)k * T_LEN + ts] : 0.0f));
  }
  // unshifted x + cond: 4-channel x 4-time quads (b128 loads, b64 LDS stores)
  for (int q = tid; q < 48 * 8; q += 256) {
    int kq = q >> 3, nq = q & 7;
    int k4 = RC + kq * 4, n4 = nq * 4;
    const float* srow = (k4 < 2 * RC) ? (xb + (size_t)(k4 - RC) * T_LEN)
                                      : (cb + (size_t)(k4 - 2 * RC) * T_LEN);
    const float* p0 = srow + t0 + n4;
    v4f r0 = *(const v4f*)(p0);
    v4f r1 = *(const v4f*)(p0 + T_LEN);
    v4f r2 = *(const v4f*)(p0 + 2 * T_LEN);
    v4f r3 = *(const v4f*)(p0 + 3 * T_LEN);
#pragma unroll
    for (int dn = 0; dn < 4; ++dn) {
      int n = n4 + dn;
      int tile = (k4 >> 5) * 2 + (n >> 4);
      int lane = (n & 15) | (((k4 >> 4) & 1) << 4);
      v4h pk = { f2h(r0[dn]), f2h(r1[dn]), f2h(r2[dn]), f2h(r3[dn]) };
      *(v4h*)(Xin + (tile * 32 + lane) * 16 + (k4 & 15)) = pk;
    }
  }
  __syncthreads();

  const int wave = tid >> 5, lane = tid & 31;
  const int fn = lane & 15, fm = (lane >> 4) * 8;

  // h = W1(128x256) @ Xin(256x32): A from global packed, no barriers in k-loop
  v8f acc0 = {}, acc1 = {};
#pragma unroll
  for (int kk = 0; kk < 8; ++kk) {
    v16h a  = load_frag_a(W1p, wave * 8 + kk);
    v16h b0 = load_frag_b(Xin, kk * 2 + 0);
    v16h b1 = load_frag_b(Xin, kk * 2 + 1);
    acc0 = wmma_f16(a, b0, acc0);
    acc1 = wmma_f16(a, b1, acc1);
  }
#pragma unroll
  for (int v = 0; v < 8; ++v) {
    int row = wave * 16 + fm + v;
    H[row * NT + fn]      = acc0[v] + hb[row];
    H[row * NT + 16 + fn] = acc1[v] + hb[row];
  }
  __syncthreads();

  // gate: channel pairs -> one b32 LDS store
  for (int id = tid; id < (RC / 2) * NT; id += 256) {
    int j2 = id >> 5, n = id & 31;
    int j = j2 * 2;
    float g0 = tanhf(H[j * NT + n]) * sigmoidf_(H[(j + RC) * NT + n]);
    float g1 = tanhf(H[(j + 1) * NT + n]) * sigmoidf_(H[(j + 1 + RC) * NT + n]);
    int tile = (j >> 5) * 2 + (n >> 4);
    int ln   = (n & 15) | (((j >> 4) & 1) << 4);
    v2h pk = { f2h(g0), f2h(g1) };
    *(v2h*)(G + (tile * 32 + ln) * 16 + (j & 15)) = pk;
  }
  __syncthreads();

  // skip_sum += skip_w(256x64) @ G + skip_b
  float* ss = skip_sum + (size_t)bt * SCH * T_LEN;
  v8f sacc[2][2];
#pragma unroll
  for (int q = 0; q < 2; ++q)
#pragma unroll
    for (int nc = 0; nc < 2; ++nc)
#pragma unroll
      for (int v = 0; v < 8; ++v) {
        int row = (wave + q * 8) * 16 + fm + v;
        float base = sb[row];
        if (!first) base += ss[(size_t)row * T_LEN + t0 + nc * 16 + fn];
        sacc[q][nc][v] = base;
      }
#pragma unroll
  for (int kk = 0; kk < 2; ++kk) {
#pragma unroll
    for (int q = 0; q < 2; ++q) {
      v16h a = load_frag_a(SWp, (wave + q * 8) * 2 + kk);
#pragma unroll
      for (int nc = 0; nc < 2; ++nc) {
        v16h bb = load_frag_b(G, kk * 2 + nc);
        sacc[q][nc] = wmma_f16(a, bb, sacc[q][nc]);
      }
    }
  }
#pragma unroll
  for (int q = 0; q < 2; ++q)
#pragma unroll
    for (int nc = 0; nc < 2; ++nc)
#pragma unroll
      for (int v = 0; v < 8; ++v) {
        int row = (wave + q * 8) * 16 + fm + v;
        ss[(size_t)row * T_LEN + t0 + nc * 16 + fn] = sacc[q][nc][v];
      }

  // x_out = res_w(64x64) @ G + res_b + x_in  (8 tiles -> one per wave)
  float* xo = x_out + (size_t)bt * RC * T_LEN;
  {
    int r0 = (wave >> 1) * 16;
    int nc = wave & 1;
    v8f racc;
#pragma unroll
    for (int v = 0; v < 8; ++v) {
      int row = r0 + fm + v;
      racc[v] = rb[row] + xb[(size_t)row * T_LEN + t0 + nc * 16 + fn];
    }
#pragma unroll
    for (int kk = 0; kk < 2; ++kk) {
      v16h a  = load_frag_a(RWp, (wave >> 1) * 2 + kk);
      v16h bb = load_frag_b(G, kk * 2 + nc);
      racc = wmma_f16(a, bb, racc);
    }
#pragma unroll
    for (int v = 0; v < 8; ++v) {
      int row = r0 + fm + v;
      xo[(size_t)row * T_LEN + t0 + nc * 16 + fn] = racc[v];
    }
  }
}

// ---------------- head ------------------------------------------------------
__global__ __launch_bounds__(256)
void head_kernel(const float* __restrict__ skip_sum,
                 const _Float16* __restrict__ O1p, const float* __restrict__ out1_b,
                 const _Float16* __restrict__ O2p, const float* __restrict__ out2_b,
                 float* __restrict__ out)
{
  __shared__ __align__(16) _Float16 Sbuf[SCH * 64];  // 32KB, B-swizzled (ntn=4), reused for s1
  __shared__ float bias1[SCH], bias2[SCH];

  const int tid = threadIdx.x;
  const int nTB = T_LEN / 64;
  const int bt = blockIdx.x / nTB;
  const int t0 = (blockIdx.x % nTB) * 64;
  const float* ss = skip_sum + (size_t)bt * SCH * T_LEN;

  // relu(skip) staging: 4-row x 4-time quads
  for (int q = tid; q < 64 * 16; q += 256) {
    int rq = q >> 4, nq = q & 15;
    int r4 = rq * 4, n4 = nq * 4;
    const float* p0 = ss + (size_t)r4 * T_LEN + t0 + n4;
    v4f c0 = *(const v4f*)(p0);
    v4f c1 = *(const v4f*)(p0 + T_LEN);
    v4f c2 = *(const v4f*)(p0 + 2 * T_LEN);
    v4f c3 = *(const v4f*)(p0 + 3 * T_LEN);
#pragma unroll
    for (int dn = 0; dn < 4; ++dn) {
      int n = n4 + dn;
      int tile = (r4 >> 5) * 4 + (n >> 4);
      int lane = (n & 15) | (((r4 >> 4) & 1) << 4);
      v4h pk = { f2h(c0[dn] > 0.0f ? c0[dn] : 0.0f),
                 f2h(c1[dn] > 0.0f ? c1[dn] : 0.0f),
                 f2h(c2[dn] > 0.0f ? c2[dn] : 0.0f),
                 f2h(c3[dn] > 0.0f ? c3[dn] : 0.0f) };
      *(v4h*)(Sbuf + (tile * 32 + lane) * 16 + (r4 & 15)) = pk;
    }
  }
  for (int i = tid; i < SCH; i += 256) { bias1[i] = out1_b[i]; bias2[i] = out2_b[i]; }
  __syncthreads();

  const int wave = tid >> 5, lane = tid & 31;
  const int fn = lane & 15, fm = (lane >> 4) * 8;

  v8f acc[2][4];
#pragma unroll
  for (int q = 0; q < 2; ++q)
#pragma unroll
    for (int nc = 0; nc < 4; ++nc) acc[q][nc] = (v8f){};

#pragma unroll
  for (int kk = 0; kk < 8; ++kk) {
#pragma unroll
    for (int q = 0; q < 2; ++q) {
      v16h a = load_frag_a(O1p, (wave + q * 8) * 8 + kk);
#pragma unroll
      for (int nc = 0; nc < 4; ++nc) {
        v16h bb = load_frag_b(Sbuf, kk * 4 + nc);
        acc[q][nc] = wmma_f16(a, bb, acc[q][nc]);
      }
    }
  }
  __syncthreads();  // all Sbuf reads done before s1 overwrite

  // s1 = relu(acc + b1) back into Sbuf; row pairs -> b32 stores
#pragma unroll
  for (int q = 0; q < 2; ++q)
#pragma unroll
    for (int nc = 0; nc < 4; ++nc)
#pragma unroll
      for (int v = 0; v < 8; v += 2) {
        int row = (wave + q * 8) * 16 + fm + v;
        float sa = acc[q][nc][v] + bias1[row];
        float sbv = acc[q][nc][v + 1] + bias1[row + 1];
        int col = nc * 16 + fn;
        int tile = (row >> 5) * 4 + (col >> 4);
        int ln   = (col & 15) | (((row >> 4) & 1) << 4);
        v2h pk = { f2h(sa > 0.0f ? sa : 0.0f), f2h(sbv > 0.0f ? sbv : 0.0f) };
        *(v2h*)(Sbuf + (tile * 32 + ln) * 16 + (row & 15)) = pk;
      }
  __syncthreads();

#pragma unroll
  for (int q = 0; q < 2; ++q)
#pragma unroll
    for (int nc = 0; nc < 4; ++nc) acc[q][nc] = (v8f){};

#pragma unroll
  for (int kk = 0; kk < 8; ++kk) {
#pragma unroll
    for (int q = 0; q < 2; ++q) {
      v16h a = load_frag_a(O2p, (wave + q * 8) * 8 + kk);
#pragma unroll
      for (int nc = 0; nc < 4; ++nc) {
        v16h bb = load_frag_b(Sbuf, kk * 4 + nc);
        acc[q][nc] = wmma_f16(a, bb, acc[q][nc]);
      }
    }
  }
  float* ob = out + (size_t)bt * OCH * T_LEN;
#pragma unroll
  for (int q = 0; q < 2; ++q)
#pragma unroll
    for (int nc = 0; nc < 4; ++nc)
#pragma unroll
      for (int v = 0; v < 8; ++v) {
        int row = (wave + q * 8) * 16 + fm + v;
        ob[(size_t)row * T_LEN + t0 + nc * 16 + fn] = acc[q][nc][v] + bias2[row];
      }
}

extern "C" void kernel_launch(void* const* d_in, const int* in_sizes, int n_in,
                              void* d_out, int out_size, void* d_ws, size_t ws_size,
                              hipStream_t stream)
{
  (void)in_sizes; (void)n_in; (void)out_size; (void)ws_size;
  const float* audio  = (const float*)d_in[0];
  const float* mfcc   = (const float*)d_in[1];
  const float* wih_f  = (const float*)d_in[2];
  const float* whh_f  = (const float*)d_in[3];
  const float* b_f    = (const float*)d_in[4];
  const float* wih_r  = (const float*)d_in[5];
  const float* whh_r  = (const float*)d_in[6];
  const float* b_r    = (const float*)d_in[7];
  const float* proj_w = (const float*)d_in[8];
  const float* proj_b = (const float*)d_in[9];
  const float* up_w   = (const float*)d_in[10];
  const float* up_b   = (const float*)d_in[11];
  const float* in_w   = (const float*)d_in[12];
  const float* in_b   = (const float*)d_in[13];
  const float* dil_w  = (const float*)d_in[14];
  const float* dil_b  = (const float*)d_in[15];
  const float* cond_w = (const float*)d_in[16];
  const float* cond_b = (const float*)d_in[17];
  const float* res_w  = (const float*)d_in[18];
  const float* res_b  = (const float*)d_in[19];
  const float* skip_w = (const float*)d_in[20];
  const float* skip_b = (const float*)d_in[21];
  const float* out1_w = (const float*)d_in[22];
  const float* out1_b = (const float*)d_in[23];
  const float* out2_w = (const float*)d_in[24];
  const float* out2_b = (const float*)d_in[25];

  float* ws = (float*)d_ws;
  size_t off = 0;
  float* hf   = ws + off; off += (size_t)2 * TF * HIDN;
  float* hr   = ws + off; off += (size_t)2 * TF * HIDN;
  float* cbuf = ws + off; off += (size_t)2 * CCH * TF;
  float* cond = ws + off; off += (size_t)2 * CCH * T_LEN;
  float* xa   = ws + off; off += (size_t)2 * RC * T_LEN;
  float* xb2  = ws + off; off += (size_t)2 * RC * T_LEN;
  float* ssum = ws + off; off += (size_t)2 * SCH * T_LEN;
  float* up_wT = ssum;  // alias: consumed before layer 0, which fully rewrites ssum

  // packed f16 weight region (A-fragment swizzled)
  _Float16* hws = (_Float16*)(ws + off);
  size_t hoff = 0;
  _Float16* W1p = hws + hoff; hoff += (size_t)NLAY * CH * KTOT;   // 16*32768
  _Float16* SWp = hws + hoff; hoff += (size_t)NLAY * SCH * RC;    // 16*16384
  _Float16* RWp = hws + hoff; hoff += (size_t)NLAY * RC * RC;     // 16*4096
  _Float16* O1p = hws + hoff; hoff += (size_t)SCH * SCH;
  _Float16* O2p = hws + hoff; hoff += (size_t)SCH * SCH;

  {
    int per_layer = CH * KTOT + SCH * RC + RC * RC;   // 53248 = 208*256
    dim3 g((per_layer + 255) / 256, NLAY);
    pack_weights_kernel<<<g, 256, 0, stream>>>(dil_w, cond_w, skip_w, res_w, W1p, SWp, RWp);
    pack_head_kernel<<<(2 * SCH * SCH + 255) / 256, 256, 0, stream>>>(out1_w, out2_w, O1p, O2p);
  }

  lstm_kernel<<<4, 1024, 0, stream>>>(mfcc, wih_f, whh_f, b_f, wih_r, whh_r, b_r, hf, hr);
  proj_kernel<<<2 * TF, 128, 0, stream>>>(hf, hr, proj_w, proj_b, cbuf);

  int ttot = CCH * CCH * 320;
  upw_transpose_kernel<<<(ttot + 255) / 256, 256, 0, stream>>>(up_w, up_wT);
  upsample_kernel<<<2 * 160, 128, 0, stream>>>(cbuf, up_wT, up_b, cond);

  int xtot = 2 * RC * T_LEN;
  xinit_kernel<<<(xtot + 255) / 256, 256, 0, stream>>>(audio, in_w, in_b, xa);

  const int DIL[NLAY] = {1, 2, 4, 8, 16, 32, 64, 128, 1, 2, 4, 8, 16, 32, 64, 128};
  float* xin = xa;
  float* xout = xb2;
  int layer_blocks = 2 * (T_LEN / NT);
  for (int i = 0; i < NLAY; ++i) {
    layer_kernel<<<layer_blocks, 256, 0, stream>>>(
        xin, xout, cond,
        W1p + (size_t)i * CH * KTOT, dil_b + (size_t)i * CH, cond_b + (size_t)i * CH,
        SWp + (size_t)i * SCH * RC,  skip_b + (size_t)i * SCH,
        RWp + (size_t)i * RC * RC,   res_b + (size_t)i * RC,
        ssum, DIL[i], (i == 0) ? 1 : 0);
    float* tmp = xin; xin = xout; xout = tmp;
  }

  int head_blocks = 2 * (T_LEN / 64);
  head_kernel<<<head_blocks, 256, 0, stream>>>(ssum, O1p, out1_b, O2p, out2_b, (float*)d_out);
}
